// Network_18056042512985
// MI455X (gfx1250) — matrix operand order repaired and checked by reading, not compiled, not run on hardware
//
#include <hip/hip_runtime.h>
#include <math.h>
#include <stdint.h>

// ---------------------------------------------------------------- types
typedef __bf16 bf16_t;
typedef bf16_t v16bf __attribute__((ext_vector_type(16)));
typedef bf16_t v8bf  __attribute__((ext_vector_type(8)));
typedef float  v8f   __attribute__((ext_vector_type(8)));

// ---------------------------------------------------------------- sizes
#define NEXD   2
#define L_IND  256
#define BBD    256
#define L_OUTD 128
#define HHD    512
#define EED    128
#define VVD    65            // V+1 for both vocabularies
#define MMD    512           // N_EX * B
#define FOURH  2048
#define NEGV   (-1000000000.0f)

// ---------------------------------------------------------------- helpers
__device__ inline bf16_t f2bf(float f) {
  union { float f; uint32_t u; } v; v.f = f;
  uint32_t r = (v.u + 0x7fffu + ((v.u >> 16) & 1u)) >> 16;
  union { uint16_t u; bf16_t b; } o; o.u = (uint16_t)r;
  return o.b;
}
__device__ inline float bf2f(bf16_t b) {
  union { uint16_t u; bf16_t b; } i; i.b = b;
  union { uint32_t u; float f; } o; o.u = ((uint32_t)i.u) << 16;
  return o.f;
}
__device__ inline float ld2f(float v)  { return v; }
__device__ inline float ld2f(bf16_t v) { return bf2f(v); }
__device__ inline float sigm(float x)  { return 1.0f / (1.0f + expf(-x)); }

// ================================================================
// WMMA GEMM:  out[m, n] = sum_k A[m, k] * Bw[n, k]   (Bw row-major N x K, bf16)
// MODE 0: LSTM pre-activation: + bih[n] + bhh[n] + Wih[n*65 + tok[m]] -> f32 outF
// MODE 1: store bf16 -> outB
// MODE 2: tanh(v + bih[n]) -> f32 outF   (bih doubles as Wb)
// Block tile BMT x 128, K-step 32, 8 waves (wave32).
//   BMT=128: wave tile 64x32 (8 accums);  BMT=64: wave tile 32x32 (4 accums)
// B tile staged into LDS with GLOBAL_LOAD_ASYNC_TO_LDS_B128 (ASYNCcnt),
// A tile staged with VALU f32->bf16 conversion; next-tile global prefetch.
// ================================================================
#define BN 128
#define BKK 32
#define LDP 40   // padded LDS row (bf16 elements), keeps 16B alignment

template <typename TA, int MODE, int BMT>
__global__ __launch_bounds__(256)
void gemm_absT(const TA* __restrict__ A, const bf16_t* __restrict__ Bw,
               float* __restrict__ outF, bf16_t* __restrict__ outB,
               int K, int lda, int ldc,
               const float* __restrict__ bih, const float* __restrict__ bhh,
               const float* __restrict__ Wih, const int* __restrict__ tok)
{
  __shared__ __align__(16) bf16_t As[BMT][LDP];
  __shared__ __align__(16) bf16_t Bs[BN][LDP];

  constexpr int TI = BMT / 32;           // M sub-tiles per wave (4 or 2)
  const int tid  = threadIdx.x;
  const int lane = tid & 31;
  const int wave = tid >> 5;
  const int waveRow = wave & 1;          // 2 wave rows
  const int waveCol = wave >> 1;         // 4 wave cols of 32
  const size_t bm0 = (size_t)blockIdx.y * BMT;
  const int    bn0 = blockIdx.x * BN;
  const int wm0 = waveRow * (BMT / 2);
  const int wn0 = waveCol * 32;

  v8f acc[TI][2] = {};

  // A loader mapping: PT elems/thread, TPR threads per 32-wide row
  constexpr int PT  = BMT / 8;           // 16 (BMT=128) or 8 (BMT=64)
  constexpr int TPR = 32 / PT;
  const int ra    = tid / TPR;
  const int cseg  = (tid % TPR) * PT;

  for (int k0 = 0; k0 < K; k0 += BKK) {
    // ---- B tile: async DMA global -> LDS (16B per lane per issue) ----
    #pragma unroll
    for (int ch = tid; ch < (BN * BKK) / 8; ch += 256) {
      const int rb = ch >> 2;            // 0..127
      const int c8 = (ch & 3) * 8;       // 0,8,16,24
      uint32_t ldsa = (uint32_t)(uintptr_t)&Bs[rb][c8];
      uint64_t ga   = (uint64_t)(uintptr_t)(Bw + (size_t)(bn0 + rb) * (size_t)K + k0 + c8);
      asm volatile("global_load_async_to_lds_b128 %0, %1, off"
                   :: "v"(ldsa), "v"(ga) : "memory");
    }
    // ---- A tile: load + convert to bf16 in LDS ----
    {
      const TA* srcA = A + (bm0 + (size_t)ra) * (size_t)lda + k0 + cseg;
      #pragma unroll
      for (int j = 0; j < PT; ++j) As[ra][cseg + j] = f2bf(ld2f(srcA[j]));
      if (k0 + BKK < K) __builtin_prefetch(srcA + BKK, 0, 1);
    }
    asm volatile("s_wait_asynccnt 0" ::: "memory");
    __syncthreads();

    // ---- build fragments per the CDNA5 16-bit WMMA VGPR layout ----
    const int halfm = lane >> 4;     // 0: lanes 0-15, 1: lanes 16-31
    const int lrow  = lane & 15;
    v16bf a[TI], b[2];
    #pragma unroll
    for (int ti = 0; ti < TI; ++ti) {
      // A 16x32: lane<16 -> K {0..7,16..23}; lane>=16 -> K {8..15,24..31}
      const bf16_t* p = &As[wm0 + 16 * ti + lrow][halfm * 8];
      v8bf lo = *(const v8bf*)p;
      v8bf hi = *(const v8bf*)(p + 16);
      a[ti] = __builtin_shufflevector(lo, hi, 0,1,2,3,4,5,6,7,8,9,10,11,12,13,14,15);
    }
    #pragma unroll
    for (int tj = 0; tj < 2; ++tj) {
      // B 32x16: lane<16 -> N=lane, K {0..7,16..23}; lane>=16 -> K {8..15,24..31}
      const bf16_t* p = &Bs[wn0 + 16 * tj + lrow][halfm * 8];
      v8bf lo = *(const v8bf*)p;
      v8bf hi = *(const v8bf*)(p + 16);
      b[tj] = __builtin_shufflevector(lo, hi, 0,1,2,3,4,5,6,7,8,9,10,11,12,13,14,15);
    }

    #pragma unroll
    for (int ti = 0; ti < TI; ++ti)
      #pragma unroll
      for (int tj = 0; tj < 2; ++tj)
        acc[ti][tj] = __builtin_amdgcn_wmma_f32_16x16x32_bf16(
            false, a[ti], false, b[tj], (short)0, acc[ti][tj], false, false);
    __syncthreads();
  }

  // ---- epilogue (C/D layout: lane<16: M=j, N=lane; lane>=16: M=j+8, N=lane-16) ----
  const int lrow2 = lane & 15;
  const int rhalf = (lane >> 4) * 8;
  #pragma unroll
  for (int ti = 0; ti < TI; ++ti)
    for (int tj = 0; tj < 2; ++tj)
      #pragma unroll
      for (int j = 0; j < 8; ++j) {
        size_t rr = bm0 + (size_t)(wm0 + 16 * ti + j + rhalf);
        int    cc = bn0 + wn0 + 16 * tj + lrow2;
        float  v  = acc[ti][tj][j];
        if (MODE == 0) {
          v += bih[cc] + bhh[cc] + Wih[cc * VVD + tok[rr]];
          outF[rr * (size_t)ldc + cc] = v;
        } else if (MODE == 1) {
          outB[rr * (size_t)ldc + cc] = f2bf(v);
        } else {
          outF[rr * (size_t)ldc + cc] = tanhf(v + bih[cc]);
        }
      }
}

// ================================================================ LSTM gates
__global__ __launch_bounds__(256)
void lstm_gates_kernel(const float* __restrict__ g, float* __restrict__ h,
                       float* __restrict__ c, bf16_t* __restrict__ hall,
                       bf16_t* __restrict__ xcat)
{
  int idx = blockIdx.x * blockDim.x + threadIdx.x;   // MMD*HHD
  int m  = idx >> 9;
  int hh = idx & (HHD - 1);
  const float* gr = g + (size_t)m * FOURH;
  float gi = gr[hh];
  float gf = gr[HHD + hh];
  float gg = gr[2 * HHD + hh];
  float go = gr[3 * HHD + hh];
  float cn = sigm(gf) * c[idx] + sigm(gi) * tanhf(gg);
  float hn = sigm(go) * tanhf(cn);
  c[idx] = cn;
  h[idx] = hn;
  if (hall) hall[idx] = f2bf(hn);
  if (xcat) xcat[(size_t)m * (2 * HHD) + hh] = f2bf(hn);
}

// ================================================================ attention
// One block per batch row nb (512 blocks). Online softmax over L=256 in tiles
// of 64: scores = HA[l,nb,:].h  (+mask), running max/sum, context accumulated
// per-thread (2 hidden dims each). Writes context bf16 into xcat[:, 512:1024].
__global__ __launch_bounds__(256)
void attention_kernel(const float* __restrict__ hd, const bf16_t* __restrict__ HA,
                      const bf16_t* __restrict__ Hall, const float* __restrict__ maskF,
                      bf16_t* __restrict__ xcat)
{
  const int nb   = blockIdx.x;
  const int tid  = threadIdx.x;
  const int lane = tid & 31;
  const int wave = tid >> 5;

  __shared__ float hv[HHD];
  __shared__ float sc[64];
  __shared__ float st[3];   // running max, running sum, tile rescale

  hv[2 * tid]     = hd[(size_t)nb * HHD + 2 * tid];
  hv[2 * tid + 1] = hd[(size_t)nb * HHD + 2 * tid + 1];
  if (tid == 0) { st[0] = -INFINITY; st[1] = 0.f; st[2] = 1.f; }
  float c0 = 0.f, c1 = 0.f;
  __syncthreads();

  for (int lt = 0; lt < L_IND / 64; ++lt) {
    const int l0 = lt * 64;
    // 8 waves x 8 rows: each wave computes 8 score dot-products.
    // Each lane handles a contiguous 16-element (32B) segment of the row.
    #pragma unroll
    for (int i = 0; i < 8; ++i) {
      int l = l0 + wave * 8 + i;
      const bf16_t* rp = HA + ((size_t)l * MMD + nb) * HHD + lane * 16;
      float s = 0.f;
      #pragma unroll
      for (int j = 0; j < 16; ++j) s += bf2f(rp[j]) * hv[lane * 16 + j];
      #pragma unroll
      for (int off = 16; off > 0; off >>= 1) s += __shfl_xor(s, off, 32);
      if (lane == 0) sc[wave * 8 + i] = s + maskF[(size_t)l * MMD + nb];
    }
    __syncthreads();
    if (tid == 0) {
      float om = st[0];
      float tm = om;
      for (int j = 0; j < 64; ++j) tm = fmaxf(tm, sc[j]);
      float scale = (om == -INFINITY) ? 0.f : expf(om - tm);
      float ss = st[1] * scale;
      for (int j = 0; j < 64; ++j) ss += expf(sc[j] - tm);
      st[0] = tm; st[1] = ss; st[2] = scale;
    }
    __syncthreads();
    const float scale = st[2];
    const float nm    = st[0];
    c0 *= scale; c1 *= scale;
    for (int j = 0; j < 64; ++j) {
      float w = expf(sc[j] - nm);
      const bf16_t* hr = Hall + (((size_t)(l0 + j)) * MMD + nb) * HHD + 2 * tid;
      c0 += w * bf2f(hr[0]);
      c1 += w * bf2f(hr[1]);
    }
    __syncthreads();
  }
  const float inv = 1.0f / st[1];
  xcat[(size_t)nb * (2 * HHD) + HHD + 2 * tid]     = f2bf(c0 * inv);
  xcat[(size_t)nb * (2 * HHD) + HHD + 2 * tid + 1] = f2bf(c1 * inv);
}

// ================================================================ scoring
__global__ __launch_bounds__(128)
void out_step_kernel(const float* __restrict__ fc, const float* __restrict__ Vw,
                     const float* __restrict__ Vb, const int* __restrict__ tgt,
                     float* __restrict__ score, float* __restrict__ act)
{
  const int b = blockIdx.x;
  const int e = threadIdx.x;
  __shared__ float mE[EED];
  __shared__ float lg[VVD];
  mE[e] = fmaxf(fc[(size_t)b * EED + e], fc[(size_t)(b + BBD) * EED + e]);
  __syncthreads();
  if (e < VVD) {
    float s = Vb[e];
    for (int k = 0; k < EED; ++k) s += mE[k] * Vw[e * EED + k];
    lg[e] = s;
  }
  __syncthreads();
  if (e == 0) {
    float mx = -INFINITY;
    for (int v = 0; v < VVD; ++v) mx = fmaxf(mx, lg[v]);
    float se = 0.f;
    for (int v = 0; v < VVD; ++v) se += expf(lg[v] - mx);
    float lse = mx + logf(se);
    int t = tgt[b];
    score[b] += (lg[t] - lse) * act[b];
    act[b] *= (t != 64) ? 1.f : 0.f;
  }
}

// ================================================================ prep / small
__global__ void f32_to_bf16_kernel(const float* __restrict__ s, bf16_t* __restrict__ d, int n) {
  int i = blockIdx.x * blockDim.x + threadIdx.x;
  if (i < n) d[i] = f2bf(s[i]);
}
__global__ void transpose_bf16_kernel(const float* __restrict__ A, bf16_t* __restrict__ At) {
  int i = blockIdx.x * blockDim.x + threadIdx.x;     // HHD*HHD
  int r = i >> 9, c = i & (HHD - 1);
  At[(size_t)c * HHD + r] = f2bf(A[(size_t)r * HHD + c]);
}
__global__ void tokE_kernel(const int* __restrict__ inp, int* __restrict__ tokE) {
  int i = blockIdx.x * blockDim.x + threadIdx.x;     // L_IND*MMD
  int l = i >> 9, m = i & (MMD - 1), n = m >> 8, b = m & 255;
  tokE[i] = inp[(size_t)n * L_IND * BBD + (size_t)l * BBD + b];
}
__global__ void tokD_kernel(const int* __restrict__ tg, int* __restrict__ tokD) {
  int i = blockIdx.x * blockDim.x + threadIdx.x;     // L_OUTD*MMD
  int t = i >> 9, m = i & (MMD - 1);
  tokD[i] = tg[t * BBD + (m & 255)];
}
__global__ void init_state_kernel(const float* __restrict__ h0, const float* __restrict__ c0,
                                  float* __restrict__ h, float* __restrict__ c) {
  int idx = blockIdx.x * blockDim.x + threadIdx.x;   // MMD*HHD
  int hh = idx & (HHD - 1);
  h[idx] = h0[hh];
  c[idx] = c0[hh];
}
__global__ void init_cd_kernel(const float* __restrict__ c0, float* __restrict__ c) {
  int idx = blockIdx.x * blockDim.x + threadIdx.x;   // MMD*HHD
  c[idx] = c0[idx & (HHD - 1)];
}
__global__ void init_score_kernel(float* __restrict__ score, float* __restrict__ act,
                                  int* __restrict__ tokS) {
  int i = blockIdx.x * blockDim.x + threadIdx.x;
  if (i < BBD) { score[i] = 0.f; act[i] = 1.f; }
  if (i < MMD) tokS[i] = 64;
}
__global__ void mask_emb_kernel(const int* __restrict__ inp, float* __restrict__ maskF,
                                int* __restrict__ embIdx) {
  int m = blockIdx.x * blockDim.x + threadIdx.x;
  if (m >= MMD) return;
  int n = m >> 8, b = m & 255;
  float act = 1.f;
  int cnt = 0;
  for (int l = 0; l < L_IND; ++l) {
    maskF[(size_t)l * MMD + m] = (act > 0.f) ? 0.f : NEGV;
    cnt += (act > 0.f) ? 1 : 0;
    act *= (inp[(size_t)n * L_IND * BBD + (size_t)l * BBD + b] != 64) ? 1.f : 0.f;
  }
  embIdx[m] = cnt - 1;
}
__global__ void emb_gather_kernel(const bf16_t* __restrict__ hall, const int* __restrict__ embIdx,
                                  float* __restrict__ emb) {
  int idx = blockIdx.x * blockDim.x + threadIdx.x;   // MMD*HHD
  int m = idx >> 9, hh = idx & (HHD - 1);
  int l = embIdx[m];
  emb[idx] = bf2f(hall[((size_t)l * MMD + m) * HHD + hh]);
}

// ================================================================ launch
extern "C" void kernel_launch(void* const* d_in, const int* in_sizes, int n_in,
                              void* d_out, int out_size, void* d_ws, size_t ws_size,
                              hipStream_t stream) {
  (void)in_sizes; (void)n_in; (void)out_size; (void)ws_size;

  const int*   inputs = (const int*)  d_in[0];
  const int*   target = (const int*)  d_in[1];
  const float* Wih_e  = (const float*)d_in[2];
  const float* Whh_e  = (const float*)d_in[3];
  const float* bih_e  = (const float*)d_in[4];
  const float* bhh_e  = (const float*)d_in[5];
  const float* h0_e   = (const float*)d_in[6];
  const float* c0_e   = (const float*)d_in[7];
  const float* Wih_d  = (const float*)d_in[8];
  const float* Whh_d  = (const float*)d_in[9];
  const float* bih_d  = (const float*)d_in[10];
  const float* bhh_d  = (const float*)d_in[11];
  const float* c0_d   = (const float*)d_in[12];
  const float* Amat   = (const float*)d_in[13];
  const float* Ww     = (const float*)d_in[14];
  const float* Wb     = (const float*)d_in[15];
  const float* Vw     = (const float*)d_in[16];
  const float* Vb     = (const float*)d_in[17];
  float* score = (float*)d_out;          // (B,) f32 — written in place

  // ---------------- workspace carve ----------------
  uint8_t* base = (uint8_t*)d_ws;
  size_t off = 0;
  auto carve = [&](size_t bytes) {
    void* p = base + off;
    off = (off + bytes + 255) & ~(size_t)255;
    return p;
  };
  bf16_t* whhE = (bf16_t*)carve((size_t)FOURH * HHD * 2);
  bf16_t* whhD = (bf16_t*)carve((size_t)FOURH * HHD * 2);
  bf16_t* Atb  = (bf16_t*)carve((size_t)HHD * HHD * 2);
  bf16_t* Wwb  = (bf16_t*)carve((size_t)EED * 2 * HHD * 2);
  bf16_t* Hall = (bf16_t*)carve((size_t)L_IND * MMD * HHD * 2);
  bf16_t* HAb  = (bf16_t*)carve((size_t)L_IND * MMD * HHD * 2);
  float*  gbuf = (float*) carve((size_t)MMD * FOURH * 4);
  float*  hE   = (float*) carve((size_t)MMD * HHD * 4);
  float*  cE   = (float*) carve((size_t)MMD * HHD * 4);
  float*  hD   = (float*) carve((size_t)MMD * HHD * 4);
  float*  cD   = (float*) carve((size_t)MMD * HHD * 4);
  float*  emb  = (float*) carve((size_t)MMD * HHD * 4);
  float*  maskF= (float*) carve((size_t)L_IND * MMD * 4);
  bf16_t* xcat = (bf16_t*)carve((size_t)MMD * 2 * HHD * 2);
  float*  fc   = (float*) carve((size_t)MMD * EED * 4);
  int*    tokE = (int*)   carve((size_t)L_IND * MMD * 4);
  int*    tokD = (int*)   carve((size_t)L_OUTD * MMD * 4);
  int*    tokS = (int*)   carve((size_t)MMD * 4);
  int*    embI = (int*)   carve((size_t)MMD * 4);
  float*  act  = (float*) carve((size_t)BBD * 4);

  const dim3 blk256(256);

  // ---------------- prep ----------------
  f32_to_bf16_kernel<<<(FOURH * HHD) / 256, blk256, 0, stream>>>(Whh_e, whhE, FOURH * HHD);
  f32_to_bf16_kernel<<<(FOURH * HHD) / 256, blk256, 0, stream>>>(Whh_d, whhD, FOURH * HHD);
  f32_to_bf16_kernel<<<(EED * 2 * HHD) / 256, blk256, 0, stream>>>(Ww, Wwb, EED * 2 * HHD);
  transpose_bf16_kernel<<<(HHD * HHD) / 256, blk256, 0, stream>>>(Amat, Atb);
  tokE_kernel<<<(L_IND * MMD) / 256, blk256, 0, stream>>>(inputs, tokE);
  tokD_kernel<<<(L_OUTD * MMD) / 256, blk256, 0, stream>>>(target, tokD);
  init_state_kernel<<<(MMD * HHD) / 256, blk256, 0, stream>>>(h0_e, c0_e, hE, cE);
  init_cd_kernel<<<(MMD * HHD) / 256, blk256, 0, stream>>>(c0_d, cD);
  init_score_kernel<<<2, blk256, 0, stream>>>(score, act, tokS);
  mask_emb_kernel<<<2, blk256, 0, stream>>>(inputs, maskF, embI);

  const dim3 gLSTM(FOURH / BN, MMD / 64);        // (16, 8)  BMT=64
  const dim3 gHA(HHD / BN, (L_IND * MMD) / 128); // (4, 1024) BMT=128
  const dim3 gFC(EED / BN, MMD / 64);            // (1, 8)  BMT=64

  // ---------------- encoder: 256 sequential steps ----------------
  for (int l = 0; l < L_IND; ++l) {
    gemm_absT<float, 0, 64><<<gLSTM, blk256, 0, stream>>>(
        hE, whhE, gbuf, nullptr, HHD, HHD, FOURH,
        bih_e, bhh_e, Wih_e, tokE + (size_t)l * MMD);
    lstm_gates_kernel<<<(MMD * HHD) / 256, blk256, 0, stream>>>(
        gbuf, hE, cE, Hall + (size_t)l * MMD * HHD, nullptr);
  }

  // ---------------- HA = Hall @ A  (one big WMMA GEMM, bf16 out) ----------------
  gemm_absT<bf16_t, 1, 128><<<gHA, blk256, 0, stream>>>(
      Hall, Atb, nullptr, HAb, HHD, HHD, HHD,
      nullptr, nullptr, nullptr, nullptr);

  // ---------------- embedding gather + decoder init LSTM step ----------------
  emb_gather_kernel<<<(MMD * HHD) / 256, blk256, 0, stream>>>(Hall, embI, emb);
  gemm_absT<float, 0, 64><<<gLSTM, blk256, 0, stream>>>(
      emb, whhD, gbuf, nullptr, HHD, HHD, FOURH,
      bih_d, bhh_d, Wih_d, tokS);
  lstm_gates_kernel<<<(MMD * HHD) / 256, blk256, 0, stream>>>(
      gbuf, hD, cD, nullptr, xcat);

  // ---------------- decoder: 128 sequential steps ----------------
  for (int t = 0; t < L_OUTD; ++t) {
    attention_kernel<<<MMD, blk256, 0, stream>>>(hD, HAb, Hall, maskF, xcat);
    gemm_absT<bf16_t, 2, 64><<<gFC, blk256, 0, stream>>>(
        xcat, Wwb, fc, nullptr, 2 * HHD, 2 * HHD, EED,
        Wb, nullptr, nullptr, nullptr);
    out_step_kernel<<<BBD, dim3(128), 0, stream>>>(
        fc, Vw, Vb, target + (size_t)t * BBD, score, act);
    gemm_absT<float, 0, 64><<<gLSTM, blk256, 0, stream>>>(
        hD, whhD, gbuf, nullptr, HHD, HHD, FOURH,
        bih_d, bhh_d, Wih_d, tokD + (size_t)t * MMD);
    lstm_gates_kernel<<<(MMD * HHD) / 256, blk256, 0, stream>>>(
        gbuf, hD, cD, nullptr, xcat);
  }
}
